// LlamaFlashAttention_SQuat_6554120094140
// MI455X (gfx1250) — compile-verified, matrix operand order checked by reading, not run on hardware
//
#include <hip/hip_runtime.h>
#include <hip/hip_bf16.h>

// ---------------- problem constants ----------------
#define NB      8      // batch
#define QL      8      // query length
#define NH      32     // query heads
#define KVH     8      // kv heads
#define HD      128    // head dim
#define HIDDEN  4096
#define TQN     3968   // quantized past length
#define FPAST   56     // full-precision past
#define LF      64     // FPAST + QL
#define STOT    4032   // TQN + LF
#define TW8     (TQN / 8)   // 496 packed int32 per d-row
#define TG32    (TQN / 32)  // 124 groups per d-row
#define PASTLEN 4024        // TQN + FPAST

#define SPLITK  8           // K-splits for weight GEMMs (4096/8 = 512 per block)
#define VSPLIT  6           // t-splits for attn@V (126 steps / 6 = 21 each)
#define VSTEP   672         // 21 * 32

typedef __attribute__((ext_vector_type(16))) __bf16 v16bf;
typedef __attribute__((ext_vector_type(4)))  __bf16 v4bf;
typedef __attribute__((ext_vector_type(2)))  __bf16 v2bf;
typedef __attribute__((ext_vector_type(8)))  float  v8f;
typedef __attribute__((ext_vector_type(4)))  float  v4f;

// Load a 16x32 bf16 WMMA operand fragment (A or B) from a row-major tile.
// ISA layout: lanes 0-15 -> K {0..7, 16..23}; lanes 16-31 -> K {8..15, 24..31}.
__device__ __forceinline__ v16bf load_frag(const __bf16* base, int stride, int lane) {
    const __bf16* p = base + (lane & 15) * stride + ((lane >> 4) << 3);
    v16bf f;
#pragma unroll
    for (int i = 0; i < 8; ++i) f[i] = p[i];
#pragma unroll
    for (int i = 0; i < 8; ++i) f[8 + i] = p[16 + i];
    return f;
}

__device__ __forceinline__ v8f wmma_bf16(v16bf a, v16bf b, v8f c) {
    return __builtin_amdgcn_wmma_f32_16x16x32_bf16(false, a, false, b, (short)0, c,
                                                   false, false);
}

// 8 contiguous f32 -> 8 bf16: two b128 global loads, two b64 LDS stores.
__device__ __forceinline__ void cvt_store8(__bf16* dst, const float* __restrict__ src) {
    v4f x0 = *(const v4f*)(src);
    v4f x1 = *(const v4f*)(src + 4);
    v4bf y0, y1;
#pragma unroll
    for (int i = 0; i < 4; ++i) { y0[i] = (__bf16)x0[i]; y1[i] = (__bf16)x1[i]; }
    *(v4bf*)(dst)     = y0;
    *(v4bf*)(dst + 4) = y1;
}

// ============================================================
// Kernel 1/8: split-K GEMM  Ypart[s][m,n] = sum_{k in split s} X[m,k]*W[n,k]
// grid = (N/64, SPLITK), 256 threads, 64x64 tile, M fixed = 64.
// ============================================================
__global__ __launch_bounds__(256)
void gemm_xwt_kernel(const float* __restrict__ X, const float* __restrict__ W,
                     float* __restrict__ Ypart, int N, int K, int Ks) {
    __shared__ __bf16 As[64][40];
    __shared__ __bf16 Bs[64][40];
    const int tid  = threadIdx.x;
    const int lane = tid & 31;
    const int wave = tid >> 5;
    const int n_base = blockIdx.x * 64;
    const int kbeg = blockIdx.y * Ks;
    const int kend = kbeg + Ks;
    const int mt  = wave & 3;        // M tile 0..3
    const int nt0 = (wave >> 2) * 2; // N tile pair {0,1} or {2,3}

    const int fm = tid >> 2;         // fill row 0..63
    const int fk = (tid & 3) * 8;    // fill k-chunk {0,8,16,24}
    const float* xrow = X + (long)fm * K + fk;
    const float* wrow = W + (long)(n_base + fm) * K + fk;

    v8f c0 = {}, c1 = {};
    for (int k0 = kbeg; k0 < kend; k0 += 32) {
        __syncthreads();
        cvt_store8(&As[fm][fk], xrow + k0);
        cvt_store8(&Bs[fm][fk], wrow + k0);
        if (k0 + 32 < kend) {                       // uniform condition
            __builtin_prefetch(xrow + k0 + 32, 0, 1);
            __builtin_prefetch(wrow + k0 + 32, 0, 1);
        }
        __syncthreads();
        v16bf a  = load_frag(&As[mt * 16][0], 40, lane);
        v16bf b0 = load_frag(&Bs[nt0 * 16][0], 40, lane);
        v16bf b1 = load_frag(&Bs[nt0 * 16 + 16][0], 40, lane);
        c0 = wmma_bf16(a, b0, c0);
        c1 = wmma_bf16(a, b1, c1);
    }
    float* Y = Ypart + (long)blockIdx.y * 64 * N;
    const int m_lo = mt * 16 + ((lane >> 4) << 3);
    const int ncol = n_base + nt0 * 16 + (lane & 15);
#pragma unroll
    for (int r = 0; r < 8; ++r) {
        Y[(long)(m_lo + r) * N + ncol]      = c0[r];
        Y[(long)(m_lo + r) * N + ncol + 16] = c1[r];
    }
}

// ============================================================
// Kernel 2/8: fixed-order partial reduction: out[i] = sum_p part[p*n + i]
// ============================================================
__global__ __launch_bounds__(256)
void reduce_sum_kernel(const float* __restrict__ part, float* __restrict__ out,
                       int n, int parts) {
    int i = blockIdx.x * 256 + threadIdx.x;
    if (i >= n) return;
    float s = 0.f;
    for (int p = 0; p < parts; ++p) s += part[(long)p * n + i];
    out[i] = s;
}

// ============================================================
// Kernel 3/8: RoPE q, RoPE+concat k_full, concat v_full -> bf16
// grid = (2048, 3) x 256
// ============================================================
__global__ __launch_bounds__(256)
void prep_rope_kernel(const float* __restrict__ qkv_q, const float* __restrict__ qkv_k,
                      const float* __restrict__ qkv_v,
                      const float* __restrict__ key_full_past,
                      const float* __restrict__ value_full_past,
                      __bf16* __restrict__ q_bf, __bf16* __restrict__ k_full,
                      __bf16* __restrict__ v_full) {
    const int idx = blockIdx.x * 256 + threadIdx.x;
    const int which = blockIdx.y;
    if (which == 0) {                       // q : B*NH*QL*64 rotate pairs
        if (idx >= NB * NH * QL * 64) return;
        int i = idx & 63, t = (idx >> 6) & 7, h = (idx >> 9) & 31, b = idx >> 14;
        const float* p = qkv_q + (long)(b * QL + t) * HIDDEN + h * HD;
        float x1 = p[i], x2 = p[i + 64];
        float inv = powf(10000.0f, -(float)i / 64.0f);
        float ang = (float)(PASTLEN + t) * inv;
        float cs = __cosf(ang), sn = __sinf(ang);
        long o = ((long)(b * NH + h) * QL + t) * HD + i;
        q_bf[o]      = (__bf16)(x1 * cs - x2 * sn);
        q_bf[o + 64] = (__bf16)(x2 * cs + x1 * sn);
    } else if (which == 1) {                // k_full : B*KVH*64*64 pairs
        if (idx >= NB * KVH * LF * 64) return;
        int i = idx & 63, j = (idx >> 6) & 63, kh = (idx >> 12) & 7, b = idx >> 15;
        float x1, x2;
        if (j < FPAST) {
            const float* p = key_full_past + ((long)(b * KVH + kh) * FPAST + j) * HD;
            x1 = p[i]; x2 = p[i + 64];
        } else {
            const float* p = qkv_k + (long)(b * QL + (j - FPAST)) * (KVH * HD) + kh * HD;
            x1 = p[i]; x2 = p[i + 64];
        }
        float inv = powf(10000.0f, -(float)i / 64.0f);
        float ang = (float)(TQN + j) * inv;
        float cs = __cosf(ang), sn = __sinf(ang);
        long o = ((long)(b * KVH + kh) * LF + j) * HD + i;
        k_full[o]      = (__bf16)(x1 * cs - x2 * sn);
        k_full[o + 64] = (__bf16)(x2 * cs + x1 * sn);
    } else {                                // v_full : B*KVH*64*128 copy
        if (idx >= NB * KVH * LF * HD) return;
        int i = idx & 127, j = (idx >> 7) & 63, kh = (idx >> 13) & 7, b = idx >> 16;
        float x = (j < FPAST)
            ? value_full_past[((long)(b * KVH + kh) * FPAST + j) * HD + i]
            : qkv_v[(long)(b * QL + (j - FPAST)) * (KVH * HD) + kh * HD + i];
        v_full[((long)(b * KVH + kh) * LF + j) * HD + i] = (__bf16)x;
    }
}

// ============================================================
// Kernel 4/8: scores vs quantized keys.
// grid = (TQN/128, B*KVH) x 256.  Dequant 4-bit K^T + fused RoPE
// into LDS, then 32 q-rows x 128 keys via WMMA.
// ============================================================
__global__ __launch_bounds__(256)
void scores_quant_kernel(const int* __restrict__ kq, const float* __restrict__ ksc,
                         const float* __restrict__ kmn, const __bf16* __restrict__ q_bf,
                         float* __restrict__ att) {
    __shared__ __bf16 Ks[128][136];   // [t_local][d]
    __shared__ __bf16 Qs[32][136];    // [m][d]
    const int tid = threadIdx.x, lane = tid & 31, wave = tid >> 5;
    const int b = blockIdx.y >> 3, kh = blockIdx.y & 7;
    const int t_base = blockIdx.x * 128;

    for (int e = tid; e < 512; e += 256) {          // Q tile: 16B vector copies
        int m = e >> 4, d8 = (e & 15) * 8;
        int hl = m >> 3, ql = m & 7;
        const __bf16* src = q_bf + ((long)(b * NH + kh * 4 + hl) * QL + ql) * HD + d8;
        *(uint4*)&Qs[m][d8] = *(const uint4*)src;
    }
    const long base_q = (long)(b * KVH + kh) * HD * TW8;
    const long base_s = (long)(b * KVH + kh) * HD * TG32;
    for (int task = tid; task < 64 * 16; task += 256) {
        int dp = task >> 4;        // d pair index 0..63
        int w  = task & 15;        // 8-key word within tile
        int t0 = t_base + w * 8;
        int g  = t0 >> 5;
        int w1 = kq[base_q + (long)dp * TW8 + (t0 >> 3)];
        int w2 = kq[base_q + (long)(dp + 64) * TW8 + (t0 >> 3)];
        float s1 = ksc[base_s + (long)dp * TG32 + g];
        float m1 = kmn[base_s + (long)dp * TG32 + g];
        float s2 = ksc[base_s + (long)(dp + 64) * TG32 + g];
        float m2 = kmn[base_s + (long)(dp + 64) * TG32 + g];
        float inv = powf(10000.0f, -(float)dp / 64.0f);
#pragma unroll
        for (int j = 0; j < 8; ++j) {
            float x1 = (float)((w1 >> (4 * j)) & 15) * s1 + m1;
            float x2 = (float)((w2 >> (4 * j)) & 15) * s2 + m2;
            float ang = (float)(t0 + j) * inv;
            float cs = __cosf(ang), sn = __sinf(ang);
            int tl = w * 8 + j;
            Ks[tl][dp]      = (__bf16)(x1 * cs - x2 * sn);
            Ks[tl][dp + 64] = (__bf16)(x2 * cs + x1 * sn);
        }
    }
    __syncthreads();
    const int mt = wave & 1, ntp = wave >> 1;       // 2 M tiles, 4 N tile pairs
    v8f c0 = {}, c1 = {};
    for (int d0 = 0; d0 < HD; d0 += 32) {
        v16bf a  = load_frag(&Qs[mt * 16][d0], 136, lane);
        v16bf b0 = load_frag(&Ks[ntp * 32][d0], 136, lane);
        v16bf b1 = load_frag(&Ks[ntp * 32 + 16][d0], 136, lane);
        c0 = wmma_bf16(a, b0, c0);
        c1 = wmma_bf16(a, b1, c1);
    }
    const float sc = 0.088388347648318447f;          // 1/sqrt(128)
    const int m_lo = mt * 16 + ((lane >> 4) << 3);
    const int n0 = t_base + ntp * 32 + (lane & 15);
#pragma unroll
    for (int r = 0; r < 8; ++r) {
        int m = m_lo + r, hl = m >> 3, ql = m & 7;
        long o = (long)((b * NH + kh * 4 + hl) * QL + ql) * STOT;
        att[o + n0]      = c0[r] * sc;
        att[o + n0 + 16] = c1[r] * sc;
    }
}

// ============================================================
// Kernel 5/8: scores vs 64 full-precision residual keys.
// grid = (B*KVH) x 256
// ============================================================
__global__ __launch_bounds__(256)
void scores_full_kernel(const __bf16* __restrict__ k_full, const __bf16* __restrict__ q_bf,
                        float* __restrict__ att) {
    __shared__ __bf16 Ks[64][136];
    __shared__ __bf16 Qs[32][136];
    const int tid = threadIdx.x, lane = tid & 31, wave = tid >> 5;
    const int b = blockIdx.x >> 3, kh = blockIdx.x & 7;

    for (int e = tid; e < 512; e += 256) {
        int m = e >> 4, d8 = (e & 15) * 8;
        int hl = m >> 3, ql = m & 7;
        const __bf16* src = q_bf + ((long)(b * NH + kh * 4 + hl) * QL + ql) * HD + d8;
        *(uint4*)&Qs[m][d8] = *(const uint4*)src;
    }
    for (int e = tid; e < 1024; e += 256) {
        int j = e >> 4, d8 = (e & 15) * 8;
        const __bf16* src = k_full + ((long)(b * KVH + kh) * LF + j) * HD + d8;
        *(uint4*)&Ks[j][d8] = *(const uint4*)src;
    }
    __syncthreads();
    const int mt = wave & 1, nt = wave >> 1;          // 2 x 4 tiles, 1 per wave
    v8f c = {};
    for (int d0 = 0; d0 < HD; d0 += 32) {
        v16bf a  = load_frag(&Qs[mt * 16][d0], 136, lane);
        v16bf bf = load_frag(&Ks[nt * 16][d0], 136, lane);
        c = wmma_bf16(a, bf, c);
    }
    const float sc = 0.088388347648318447f;
    const int m_lo = mt * 16 + ((lane >> 4) << 3);
    const int n0 = TQN + nt * 16 + (lane & 15);
#pragma unroll
    for (int r = 0; r < 8; ++r) {
        int m = m_lo + r, hl = m >> 3, ql = m & 7;
        long o = (long)((b * NH + kh * 4 + hl) * QL + ql) * STOT;
        att[o + n0] = c[r] * sc;
    }
}

// ============================================================
// Kernel 6/8: softmax over 4032, emit bf16 probabilities.
// grid = B*NH*QL = 2048 rows, 256 threads.
// ============================================================
__global__ __launch_bounds__(256)
void softmax_kernel(const float* __restrict__ att, __bf16* __restrict__ att_bf) {
    __shared__ float red[256];
    const int tid = threadIdx.x;
    const float* p = att + (long)blockIdx.x * STOT;
    float mx = -3.4e38f;
    for (int i = tid; i < STOT; i += 256) mx = fmaxf(mx, p[i]);
    red[tid] = mx; __syncthreads();
    for (int s = 128; s; s >>= 1) {
        if (tid < s) red[tid] = fmaxf(red[tid], red[tid + s]);
        __syncthreads();
    }
    mx = red[0]; __syncthreads();
    float sum = 0.f;
    for (int i = tid; i < STOT; i += 256) sum += __expf(p[i] - mx);
    red[tid] = sum; __syncthreads();
    for (int s = 128; s; s >>= 1) {
        if (tid < s) red[tid] += red[tid + s];
        __syncthreads();
    }
    const float inv = 1.0f / red[0];
    __bf16* q = att_bf + (long)blockIdx.x * STOT;
    for (int i = tid; i < STOT; i += 256) q[i] = (__bf16)(__expf(p[i] - mx) * inv);
}

// ============================================================
// Kernel 7/8: out_part = attn @ V over one t-split.
// grid = (B*KVH, VSPLIT) x 256.  B operand staged transposed in LDS.
// ============================================================
__global__ __launch_bounds__(256)
void attn_v_kernel(const int* __restrict__ vq, const float* __restrict__ vsc,
                   const float* __restrict__ vmn, const __bf16* __restrict__ v_full,
                   const __bf16* __restrict__ att_bf, float* __restrict__ out_part) {
    __shared__ __bf16 Vs[128][40];    // [d][t_local], t-step 32
    const int tid = threadIdx.x, lane = tid & 31, wave = tid >> 5;
    const int b = blockIdx.x >> 3, kh = blockIdx.x & 7;
    const int mt = wave & 1, ntp = wave >> 1;
    const int tb = blockIdx.y * VSTEP;

    const int m_row = mt * 16 + (lane & 15);
    const int hlA = m_row >> 3, qlA = m_row & 7;
    const __bf16* Arow =
        att_bf + (long)((b * NH + kh * 4 + hlA) * QL + qlA) * STOT;
    const int kb = (lane >> 4) << 3;

    v8f c0 = {}, c1 = {};
    for (int t0 = tb; t0 < tb + VSTEP; t0 += 32) {
        __syncthreads();
        if (t0 < TQN) {
            // one task per thread: (t-pair, 8-d word) -> paired b32 LDS stores
            int tp = tid >> 4, w = tid & 15;
            int tl = tp * 2;
            long rb0 = (long)(b * KVH + kh) * TQN + (t0 + tl);
            int word0 = vq[rb0 * 16 + w];
            int word1 = vq[(rb0 + 1) * 16 + w];
            float s0 = vsc[rb0 * 4 + (w >> 2)],       mn0 = vmn[rb0 * 4 + (w >> 2)];
            float s1 = vsc[(rb0 + 1) * 4 + (w >> 2)], mn1 = vmn[(rb0 + 1) * 4 + (w >> 2)];
#pragma unroll
            for (int j = 0; j < 8; ++j) {
                int d = w * 8 + j;
                v2bf pr;
                pr[0] = (__bf16)((float)((word0 >> (4 * j)) & 15) * s0 + mn0);
                pr[1] = (__bf16)((float)((word1 >> (4 * j)) & 15) * s1 + mn1);
                *(v2bf*)&Vs[d][tl] = pr;
            }
        } else {
            int jb = t0 - TQN;
            for (int e = tid; e < 32 * 128; e += 256) {
                int tl = e >> 7, d = e & 127;
                Vs[d][tl] = v_full[((long)(b * KVH + kh) * LF + jb + tl) * HD + d];
            }
        }
        __syncthreads();
        v16bf a;
        const __bf16* ap = Arow + t0 + kb;
#pragma unroll
        for (int i = 0; i < 8; ++i) a[i] = ap[i];
#pragma unroll
        for (int i = 0; i < 8; ++i) a[8 + i] = ap[16 + i];
        v16bf b0 = load_frag(&Vs[ntp * 32][0], 40, lane);
        v16bf b1 = load_frag(&Vs[ntp * 32 + 16][0], 40, lane);
        c0 = wmma_bf16(a, b0, c0);
        c1 = wmma_bf16(a, b1, c1);
    }
    float* out = out_part + (long)blockIdx.y * 64 * HIDDEN;
    const int m_lo = mt * 16 + kb;
    const int d0 = ntp * 32 + (lane & 15);
#pragma unroll
    for (int r = 0; r < 8; ++r) {
        int m = m_lo + r, hl = m >> 3, ql = m & 7;
        long o = (long)(b * QL + ql) * HIDDEN + (kh * 4 + hl) * HD;
        out[o + d0]      = c0[r];
        out[o + d0 + 16] = c1[r];
    }
}

// ============================================================
// launch
// ============================================================
extern "C" void kernel_launch(void* const* d_in, const int* in_sizes, int n_in,
                              void* d_out, int out_size, void* d_ws, size_t ws_size,
                              hipStream_t stream) {
    const float* hidden = (const float*)d_in[0];
    const float* Wq = (const float*)d_in[1];
    const float* Wk = (const float*)d_in[2];
    const float* Wv = (const float*)d_in[3];
    const float* Wo = (const float*)d_in[4];
    const int*   kq  = (const int*)d_in[5];
    const float* ksc = (const float*)d_in[6];
    const float* kmn = (const float*)d_in[7];
    const float* kfp = (const float*)d_in[8];
    const int*   vq  = (const int*)d_in[9];
    const float* vsc = (const float*)d_in[10];
    const float* vmn = (const float*)d_in[11];
    const float* vfp = (const float*)d_in[12];

    char* ws = (char*)d_ws;
    size_t off = 0;
    float*  qkv_q  = (float*)(ws + off);  off += (size_t)64 * HIDDEN * 4;      // 1 MB
    float*  qkv_k  = (float*)(ws + off);  off += (size_t)64 * KVH * HD * 4;    // 256 KB
    float*  qkv_v  = (float*)(ws + off);  off += (size_t)64 * KVH * HD * 4;
    __bf16* q_bf   = (__bf16*)(ws + off); off += (size_t)NB * NH * QL * HD * 2;
    __bf16* k_full = (__bf16*)(ws + off); off += (size_t)NB * KVH * LF * HD * 2;
    __bf16* v_full = (__bf16*)(ws + off); off += (size_t)NB * KVH * LF * HD * 2;
    float*  att    = (float*)(ws + off);  off += (size_t)NB * NH * QL * STOT * 4;
    __bf16* att_bf = (__bf16*)(ws + off); off += (size_t)NB * NH * QL * STOT * 2;
    float*  out_pre = (float*)(ws + off); off += (size_t)64 * HIDDEN * 4;
    float*  gpart  = (float*)(ws + off);  off += (size_t)SPLITK * 64 * HIDDEN * 4; // 8 MB
    float*  vpart  = (float*)(ws + off);  off += (size_t)VSPLIT * 64 * HIDDEN * 4; // 6 MB

    const int nQ = 64 * HIDDEN;          // 262144
    const int nKV = 64 * KVH * HD;       // 65536

    // 1. QKV projections (split-K + reduce)
    gemm_xwt_kernel<<<dim3(HIDDEN / 64, SPLITK), 256, 0, stream>>>(
        hidden, Wq, gpart, HIDDEN, HIDDEN, HIDDEN / SPLITK);
    reduce_sum_kernel<<<(nQ + 255) / 256, 256, 0, stream>>>(gpart, qkv_q, nQ, SPLITK);
    gemm_xwt_kernel<<<dim3((KVH * HD) / 64, SPLITK), 256, 0, stream>>>(
        hidden, Wk, gpart, KVH * HD, HIDDEN, HIDDEN / SPLITK);
    reduce_sum_kernel<<<(nKV + 255) / 256, 256, 0, stream>>>(gpart, qkv_k, nKV, SPLITK);
    gemm_xwt_kernel<<<dim3((KVH * HD) / 64, SPLITK), 256, 0, stream>>>(
        hidden, Wv, gpart, KVH * HD, HIDDEN, HIDDEN / SPLITK);
    reduce_sum_kernel<<<(nKV + 255) / 256, 256, 0, stream>>>(gpart, qkv_v, nKV, SPLITK);

    // 2. RoPE + concat + bf16 pack
    prep_rope_kernel<<<dim3(2048, 3), 256, 0, stream>>>(qkv_q, qkv_k, qkv_v, kfp, vfp,
                                                        q_bf, k_full, v_full);

    // 3/4. attention scores
    scores_quant_kernel<<<dim3(TQN / 128, NB * KVH), 256, 0, stream>>>(kq, ksc, kmn, q_bf, att);
    scores_full_kernel<<<NB * KVH, 256, 0, stream>>>(k_full, q_bf, att);

    // 5. softmax -> bf16 probabilities
    softmax_kernel<<<NB * NH * QL, 256, 0, stream>>>(att, att_bf);

    // 6. attn @ V  (t-split + reduce)
    attn_v_kernel<<<dim3(NB * KVH, VSPLIT), 256, 0, stream>>>(vq, vsc, vmn, v_full,
                                                              att_bf, vpart);
    reduce_sum_kernel<<<(nQ + 255) / 256, 256, 0, stream>>>(vpart, out_pre, nQ, VSPLIT);

    // 7. output projection (split-K + reduce into d_out)
    gemm_xwt_kernel<<<dim3(HIDDEN / 64, SPLITK), 256, 0, stream>>>(
        out_pre, Wo, gpart, HIDDEN, HIDDEN, HIDDEN / SPLITK);
    reduce_sum_kernel<<<(nQ + 255) / 256, 256, 0, stream>>>(gpart, (float*)d_out, nQ, SPLITK);
}
